// CustomGraphNet_45964740002143
// MI455X (gfx1250) — matrix operand order repaired and checked by reading, not compiled, not run on hardware
//
#include <hip/hip_runtime.h>
#include <hip/hip_bf16.h>
#include <math.h>

// ---------------------------------------------------------------------------
// CDNA5 (gfx1250) fused GNN. wave32, v_wmma_f32_16x16x32_bf16 for all GEMMs.
// ---------------------------------------------------------------------------

typedef __bf16 bf16_t;
typedef __attribute__((ext_vector_type(16))) __bf16 v16bf;
typedef __attribute__((ext_vector_type(8)))  __bf16 v8bf;
typedef __attribute__((ext_vector_type(4)))  __bf16 v4bf;
typedef __attribute__((ext_vector_type(8)))  float  v8f;
typedef __attribute__((ext_vector_type(4)))  float  v4f;

#define WAVES 2          // waves (16-row tiles) per block
#define HSTR  136        // bf16 elements per row of a 16x128 LDS tile (bank spread)

struct FwdW {
    const bf16_t *w_in, *w_l1, *w_l2, *w_out;   // packed B-fragment layout
    const float  *b_in, *b_l1, *b_l2, *b_out;
    const float  *ln_s, *ln_b;
};

// ---------------------------------------------------------------------------
// Weight packing: W[K,Nc] f32 row-major -> bf16 B-fragments.
// Fragment order: tile = kt*NT + nt; within a tile 32 lanes x 16 halves:
//   n = nt*16 + (lane&15);  k = kt*32 + ((h&8)<<1) + ((lane>>4)<<3) + (h&7)
// so each lane consumes one contiguous 32-byte v16bf per WMMA.
// ---------------------------------------------------------------------------
__global__ void k_pack(const float* __restrict__ W, bf16_t* __restrict__ out,
                       int K, int Nc, int total) {
    int t = blockIdx.x * blockDim.x + threadIdx.x;
    if (t >= total) return;
    int NT   = (Nc + 15) >> 4;
    int h    = t & 15;
    int lane = (t >> 4) & 31;
    int tile = t >> 9;
    int kt = tile / NT, nt = tile % NT;
    int k = kt * 32 + ((h & 8) << 1) + ((lane >> 4) << 3) + (h & 7);
    int n = nt * 16 + (lane & 15);
    float v = (k < K && n < Nc) ? W[(size_t)k * Nc + n] : 0.0f;
    out[t] = (bf16_t)v;
}

// ---------------------------------------------------------------------------
// A-fragment from LDS tile (bf16 row-major, 16 rows, stride elements).
// 16-bit A 16x32 layout: lane m=lane&15; lanes<16: halves0-7->K0-7, 8-15->K16-23;
// lanes>=16: halves0-7->K8-15, 8-15->K24-31  => two contiguous 8-half runs.
// ---------------------------------------------------------------------------
__device__ __forceinline__ v16bf load_a_frag(const bf16_t* tile, int stride, int ks) {
    int lane = threadIdx.x & 31;
    int m = lane & 15;
    const bf16_t* p = tile + m * stride + ks * 32 + ((lane & 16) >> 1);
    v8bf lo = *(const v8bf*)p;
    v8bf hi = *(const v8bf*)(p + 16);
    v16bf a;
#pragma unroll
    for (int i = 0; i < 8; ++i) { a[i] = lo[i]; a[i + 8] = hi[i]; }
    return a;
}

// 16x[K] (LDS, bf16) x packed W -> 16x128, +bias, ReLU, optional +=residual
// (residual source == dTile), result stored bf16 into dTile.
template<int KT, bool RES>
__device__ __forceinline__ void gemm_relu_to_lds(const bf16_t* aTile, int aStride,
                                                 const bf16_t* __restrict__ Wp,
                                                 const float* __restrict__ bias,
                                                 bf16_t* dTile) {
    int lane = threadIdx.x & 31;
    int n = lane & 15;
    v8f acc[8];
#pragma unroll
    for (int nt = 0; nt < 8; ++nt) acc[nt] = (v8f)0.0f;
#pragma unroll
    for (int ks = 0; ks < KT; ++ks) {
        v16bf a = load_a_frag(aTile, aStride, ks);
#pragma unroll
        for (int nt = 0; nt < 8; ++nt) {
            v16bf b = *(const v16bf*)(Wp + (((size_t)(ks * 8 + nt)) << 9) + (lane << 4));
            acc[nt] = __builtin_amdgcn_wmma_f32_16x16x32_bf16(
                false, a, false, b, (short)0, acc[nt], false, false);
        }
    }
    __builtin_amdgcn_wave_barrier();
#pragma unroll
    for (int nt = 0; nt < 8; ++nt) {
        float bv = bias[nt * 16 + n];
#pragma unroll
        for (int r = 0; r < 8; ++r) {
            int m = (lane & 16) ? (r + 8) : r;
            float v = fmaxf(acc[nt][r] + bv, 0.0f);
            if (RES) v += (float)dTile[m * HSTR + nt * 16 + n];
            dTile[m * HSTR + nt * 16 + n] = (bf16_t)v;
        }
    }
    __builtin_amdgcn_wave_barrier();
}

// Final 128->128 layer: +bias, optional LayerNorm, optional += global residual,
// f32 store to gout[row*128+col].
template<bool LN, bool GRES>
__device__ __forceinline__ void gemm_final128(const bf16_t* aTile,
                                              const bf16_t* __restrict__ Wp,
                                              const float* __restrict__ bias,
                                              const float* __restrict__ lnS,
                                              const float* __restrict__ lnB,
                                              const float* __restrict__ gres,
                                              float* __restrict__ gout,
                                              int rowbase, int nrows) {
    int lane = threadIdx.x & 31;
    int n = lane & 15;
    v8f acc[8];
#pragma unroll
    for (int nt = 0; nt < 8; ++nt) acc[nt] = (v8f)0.0f;
#pragma unroll
    for (int ks = 0; ks < 4; ++ks) {
        v16bf a = load_a_frag(aTile, HSTR, ks);
#pragma unroll
        for (int nt = 0; nt < 8; ++nt) {
            v16bf b = *(const v16bf*)(Wp + (((size_t)(ks * 8 + nt)) << 9) + (lane << 4));
            acc[nt] = __builtin_amdgcn_wmma_f32_16x16x32_bf16(
                false, a, false, b, (short)0, acc[nt], false, false);
        }
    }
#pragma unroll
    for (int nt = 0; nt < 8; ++nt) {
        float bv = bias[nt * 16 + n];
#pragma unroll
        for (int r = 0; r < 8; ++r) acc[nt][r] += bv;
    }
    if (LN) {
#pragma unroll
        for (int r = 0; r < 8; ++r) {
            float p = 0.0f, q = 0.0f;
#pragma unroll
            for (int nt = 0; nt < 8; ++nt) { float v = acc[nt][r]; p += v; q += v * v; }
#pragma unroll
            for (int msk = 8; msk >= 1; msk >>= 1) {
                p += __shfl_xor(p, msk, 32);
                q += __shfl_xor(q, msk, 32);
            }
            float mean = p * (1.0f / 128.0f);
            float var  = q * (1.0f / 128.0f) - mean * mean;
            float inv  = rsqrtf(var + 1e-5f);
#pragma unroll
            for (int nt = 0; nt < 8; ++nt) acc[nt][r] = (acc[nt][r] - mean) * inv;
        }
#pragma unroll
        for (int nt = 0; nt < 8; ++nt) {
            float sc = lnS[nt * 16 + n], bi = lnB[nt * 16 + n];
#pragma unroll
            for (int r = 0; r < 8; ++r) acc[nt][r] = acc[nt][r] * sc + bi;
        }
    }
#pragma unroll
    for (int nt = 0; nt < 8; ++nt) {
#pragma unroll
        for (int r = 0; r < 8; ++r) {
            int m = (lane & 16) ? (r + 8) : r;
            int row = rowbase + m;
            if (row < nrows) {
                int col = nt * 16 + n;
                float v = acc[nt][r];
                if (GRES) v += gres[(size_t)row * 128 + col];
                gout[(size_t)row * 128 + col] = v;
            }
        }
    }
}

// Decoder output layer: 128 -> 3 (one N-tile, cols 0..2 stored)
__device__ __forceinline__ void gemm_final_out3(const bf16_t* aTile,
                                                const bf16_t* __restrict__ Wp,
                                                const float* __restrict__ bias,
                                                float* __restrict__ gout,
                                                int rowbase, int nrows) {
    int lane = threadIdx.x & 31;
    int n = lane & 15;
    v8f acc = (v8f)0.0f;
#pragma unroll
    for (int ks = 0; ks < 4; ++ks) {
        v16bf a = load_a_frag(aTile, HSTR, ks);
        v16bf b = *(const v16bf*)(Wp + (((size_t)ks) << 9) + (lane << 4));
        acc = __builtin_amdgcn_wmma_f32_16x16x32_bf16(
            false, a, false, b, (short)0, acc, false, false);
    }
    float bv = (n < 3) ? bias[n] : 0.0f;
#pragma unroll
    for (int r = 0; r < 8; ++r) {
        int m = (lane & 16) ? (r + 8) : r;
        int row = rowbase + m;
        if (row < nrows && n < 3) gout[(size_t)row * 3 + n] = acc[r] + bv;
    }
}

// copy 64 f32 -> 64 bf16 (one lane-half of one row segment)
__device__ __forceinline__ void fill64(bf16_t* dst, const float* src) {
#pragma unroll
    for (int c = 0; c < 64; c += 4) {
        v4f f = *(const v4f*)(src + c);
        v4bf b; b.x = (bf16_t)f.x; b.y = (bf16_t)f.y; b.z = (bf16_t)f.z; b.w = (bf16_t)f.w;
        *(v4bf*)(dst + c) = b;
    }
}

// ---------------------------------------------------------------------------
// Fused fwdnet kernels (one wave == one 16-row tile)
// ---------------------------------------------------------------------------
template<int DIN>
__global__ void __launch_bounds__(64) k_encode(const float* __restrict__ in,
                                               float* __restrict__ gout,
                                               FwdW f, int nrows) {
    constexpr int KP = 40;   // K padded to 32, stride padded for banks
    __shared__ __align__(16) bf16_t sIn[WAVES][16 * KP];
    __shared__ __align__(16) bf16_t sH [WAVES][16 * HSTR];
    __shared__ __align__(16) bf16_t sR [WAVES][16 * HSTR];
    int wave = threadIdx.x >> 5, lane = threadIdx.x & 31;
    int ntiles = (nrows + 15) >> 4;
    int tile = blockIdx.x * WAVES + wave; if (tile >= ntiles) tile = ntiles - 1;
    int base = tile * 16;
    bf16_t *aIn = sIn[wave], *h = sH[wave], *r = sR[wave];
    if (lane < 16) {
        int row = base + lane; if (row >= nrows) row = nrows - 1;
        const float* s = in + (size_t)row * DIN;
#pragma unroll
        for (int k = 0; k < 32; ++k) aIn[lane * KP + k] = (bf16_t)((k < DIN) ? s[k] : 0.0f);
    }
    __builtin_amdgcn_wave_barrier();
    gemm_relu_to_lds<1, false>(aIn, KP,   f.w_in,  f.b_in,  h);
    gemm_relu_to_lds<4, false>(h,   HSTR, f.w_l1,  f.b_l1,  r);
    gemm_relu_to_lds<4, true >(r,   HSTR, f.w_l2,  f.b_l2,  h);
    gemm_final128<true, false>(h, f.w_out, f.b_out, f.ln_s, f.ln_b,
                               nullptr, gout, base, nrows);
}

__global__ void __launch_bounds__(64) k_edge_proc(const int* __restrict__ srcIdx,
                                                  const int* __restrict__ dstIdx,
                                                  const float* __restrict__ nodeLat,
                                                  float* __restrict__ edgeLat,
                                                  FwdW f, int E) {
    constexpr int KP = 392;  // 384 padded
    __shared__ __align__(16) bf16_t sIn[WAVES][16 * KP];
    __shared__ __align__(16) bf16_t sH [WAVES][16 * HSTR];
    __shared__ __align__(16) bf16_t sR [WAVES][16 * HSTR];
    int wave = threadIdx.x >> 5, lane = threadIdx.x & 31;
    int ntiles = (E + 15) >> 4;
    int tile = blockIdx.x * WAVES + wave; if (tile >= ntiles) tile = ntiles - 1;
    int base = tile * 16;
    bf16_t *aIn = sIn[wave], *h = sH[wave], *r = sR[wave];

    int r2 = lane >> 1, half = lane & 1;
    int e = base + r2; if (e >= E) e = E - 1;
    int di = dstIdx[e], si = srcIdx[e];
    bf16_t* row = aIn + r2 * KP + half * 64;
    fill64(row,       nodeLat + (size_t)di * 128 + half * 64);   // x_i
    fill64(row + 128, nodeLat + (size_t)si * 128 + half * 64);   // x_j
    fill64(row + 256, edgeLat + (size_t)e  * 128 + half * 64);   // edge latent
    __builtin_amdgcn_wave_barrier();
    gemm_relu_to_lds<12, false>(aIn, KP,   f.w_in, f.b_in, h);
    gemm_relu_to_lds<4,  false>(h,   HSTR, f.w_l1, f.b_l1, r);
    gemm_relu_to_lds<4,  true >(r,   HSTR, f.w_l2, f.b_l2, h);
    // new_edge = edge_lat + LN(lin_out(h))  (in-place: each tile owns its rows)
    gemm_final128<true, true>(h, f.w_out, f.b_out, f.ln_s, f.ln_b,
                              edgeLat, edgeLat, base, E);
}

__global__ void __launch_bounds__(64) k_node_proc(float* __restrict__ nodeLat,
                                                  const float* __restrict__ agg,
                                                  FwdW f, int N) {
    constexpr int KP = 264;  // 256 padded
    __shared__ __align__(16) bf16_t sIn[WAVES][16 * KP];
    __shared__ __align__(16) bf16_t sH [WAVES][16 * HSTR];
    __shared__ __align__(16) bf16_t sR [WAVES][16 * HSTR];
    int wave = threadIdx.x >> 5, lane = threadIdx.x & 31;
    int ntiles = (N + 15) >> 4;
    int tile = blockIdx.x * WAVES + wave; if (tile >= ntiles) tile = ntiles - 1;
    int base = tile * 16;
    bf16_t *aIn = sIn[wave], *h = sH[wave], *r = sR[wave];

    int r2 = lane >> 1, half = lane & 1;
    int row = base + r2; if (row >= N) row = N - 1;
    bf16_t* dr = aIn + r2 * KP + half * 64;
    fill64(dr,       nodeLat + (size_t)row * 128 + half * 64);
    fill64(dr + 128, agg     + (size_t)row * 128 + half * 64);
    __builtin_amdgcn_wave_barrier();
    gemm_relu_to_lds<8, false>(aIn, KP,   f.w_in, f.b_in, h);
    gemm_relu_to_lds<4, false>(h,   HSTR, f.w_l1, f.b_l1, r);
    gemm_relu_to_lds<4, true >(r,   HSTR, f.w_l2, f.b_l2, h);
    gemm_final128<true, true>(h, f.w_out, f.b_out, f.ln_s, f.ln_b,
                              nodeLat, nodeLat, base, N);
}

__global__ void __launch_bounds__(64) k_decoder(const float* __restrict__ nodeLat,
                                                float* __restrict__ gout,
                                                FwdW f, int N) {
    constexpr int KP = HSTR;
    __shared__ __align__(16) bf16_t sIn[WAVES][16 * KP];
    __shared__ __align__(16) bf16_t sH [WAVES][16 * HSTR];
    __shared__ __align__(16) bf16_t sR [WAVES][16 * HSTR];
    int wave = threadIdx.x >> 5, lane = threadIdx.x & 31;
    int ntiles = (N + 15) >> 4;
    int tile = blockIdx.x * WAVES + wave; if (tile >= ntiles) tile = ntiles - 1;
    int base = tile * 16;
    bf16_t *aIn = sIn[wave], *h = sH[wave], *r = sR[wave];

    int r2 = lane >> 1, half = lane & 1;
    int row = base + r2; if (row >= N) row = N - 1;
    fill64(aIn + r2 * KP + half * 64, nodeLat + (size_t)row * 128 + half * 64);
    __builtin_amdgcn_wave_barrier();
    gemm_relu_to_lds<4, false>(aIn, KP,   f.w_in, f.b_in, h);
    gemm_relu_to_lds<4, false>(h,   HSTR, f.w_l1, f.b_l1, r);
    gemm_relu_to_lds<4, true >(r,   HSTR, f.w_l2, f.b_l2, h);
    gemm_final_out3(h, f.w_out, f.b_out, gout, base, N);
}

// ---------------------------------------------------------------------------
// Attention + segment softmax + scatter aggregation
// ---------------------------------------------------------------------------
__global__ void k_attn(const float* __restrict__ edgeLat, const float* __restrict__ w,
                       const float* __restrict__ b, float* __restrict__ score, int E) {
    long gid = (long)blockIdx.x * blockDim.x + threadIdx.x;
    int e = (int)(gid >> 5), lane = (int)(gid & 31);
    if (e >= E) return;
    v4f x  = *(const v4f*)(edgeLat + (size_t)e * 128 + lane * 4);
    v4f wv = *(const v4f*)(w + lane * 4);
    float p = x.x * wv.x + x.y * wv.y + x.z * wv.z + x.w * wv.w;
#pragma unroll
    for (int m = 16; m >= 1; m >>= 1) p += __shfl_xor(p, m, 32);
    if (lane == 0) score[e] = fmaxf(p + b[0], 0.0f);
}

__global__ void k_zero(float* __restrict__ p, size_t n) {
    size_t i = (size_t)blockIdx.x * blockDim.x + threadIdx.x;
    size_t stride = (size_t)gridDim.x * blockDim.x;
    for (; i < n; i += stride) p[i] = 0.0f;
}

__global__ void k_smax(const int* __restrict__ dst, const float* __restrict__ score,
                       float* __restrict__ m, int E) {
    int e = blockIdx.x * blockDim.x + threadIdx.x;
    if (e >= E) return;
    atomicMax((int*)&m[dst[e]], __float_as_int(score[e]));   // scores >= 0
}

__global__ void k_expsum(const int* __restrict__ dst, const float* __restrict__ score,
                         const float* __restrict__ m, float* __restrict__ ex,
                         float* __restrict__ s, int E) {
    int e = blockIdx.x * blockDim.x + threadIdx.x;
    if (e >= E) return;
    int d = dst[e];
    float v = __expf(score[e] - m[d]);
    ex[e] = v;
    atomicAdd(&s[d], v);
}

__global__ void k_agg(const int* __restrict__ dst, const float* __restrict__ ex,
                      const float* __restrict__ s, const float* __restrict__ edgeLat,
                      float* __restrict__ agg, int E) {
    long gid = (long)blockIdx.x * blockDim.x + threadIdx.x;
    int e = (int)(gid >> 5), lane = (int)(gid & 31);
    if (e >= E) return;
    int d = dst[e];
    float alpha = ex[e] / s[d];
    const float* src = edgeLat + (size_t)e * 128 + lane * 4;
    float* dg = agg + (size_t)d * 128 + lane * 4;
#pragma unroll
    for (int c = 0; c < 4; ++c) atomicAdd(dg + c, alpha * src[c]);
}

// ---------------------------------------------------------------------------
// Host orchestration.
// d_in flattening (setup_inputs dict insertion order, depth-first):
//  0..9  node_enc  [w_in,b_in, l1w,l1b, l2w,l2b, w_out,b_out, ln_s,ln_b]
// 10..19 edge_enc
// 20..41 layers[0]: edge_proc(10), node_proc(10), attn.w, attn.b
// 42..63 layers[1]
// 64..71 decoder (no LN)
// 72 x [N,13], 73 edge_attr [E,6], 74 edge_index [2,E] int32
// ---------------------------------------------------------------------------
extern "C" void kernel_launch(void* const* d_in, const int* in_sizes, int n_in,
                              void* d_out, int out_size, void* d_ws, size_t ws_size,
                              hipStream_t stream) {
    (void)n_in; (void)out_size; (void)ws_size;
    const int NODE_IN = 13, EDGE_IN = 6;
    const int N = in_sizes[72] / NODE_IN;
    const int E = in_sizes[73] / EDGE_IN;
    auto P = [&](int i) { return (const float*)d_in[i]; };
    const int* ei = (const int*)d_in[74];
    const int* srcIdx = ei;
    const int* dstIdx = ei + E;

    // workspace carve-up
    char* ws = (char*)d_ws;
    size_t off = 0;
    auto ALLOC = [&](size_t bytes) -> void* {
        void* p = ws + off;
        off = (off + bytes + 255) & ~(size_t)255;
        return p;
    };
    float* edgeLat = (float*)ALLOC((size_t)E * 128 * 4);
    float* nodeLat = (float*)ALLOC((size_t)N * 128 * 4);
    float* agg     = (float*)ALLOC((size_t)N * 128 * 4);
    float* mbuf    = (float*)ALLOC((size_t)N * 4);
    float* sbuf    = (float*)ALLOC((size_t)N * 4);
    float* score   = (float*)ALLOC((size_t)E * 4);
    float* ex      = (float*)ALLOC((size_t)E * 4);
    bf16_t* packBase = (bf16_t*)ALLOC((size_t)2 * 1024 * 1024);
    bf16_t* pw = packBase;

    auto PACK = [&](const float* W, int K, int Nc) -> const bf16_t* {
        int KT = (K + 31) / 32, NT = (Nc + 15) / 16;
        int total = KT * NT * 512;
        bf16_t* o = pw; pw += total;
        k_pack<<<dim3((total + 255) / 256), dim3(256), 0, stream>>>(W, o, K, Nc, total);
        return o;
    };
    auto MKF = [&](int base, int din, int dout, bool ln) -> FwdW {
        FwdW f;
        f.w_in  = PACK(P(base + 0), din, 128); f.b_in  = P(base + 1);
        f.w_l1  = PACK(P(base + 2), 128, 128); f.b_l1  = P(base + 3);
        f.w_l2  = PACK(P(base + 4), 128, 128); f.b_l2  = P(base + 5);
        f.w_out = PACK(P(base + 6), 128, dout); f.b_out = P(base + 7);
        f.ln_s = ln ? P(base + 8) : nullptr;
        f.ln_b = ln ? P(base + 9) : nullptr;
        return f;
    };

    FwdW fNodeEnc = MKF(0, 13, 128, true);
    FwdW fEdgeEnc = MKF(10, 6, 128, true);
    FwdW fEdgeProc[2] = { MKF(20, 384, 128, true), MKF(42, 384, 128, true) };
    FwdW fNodeProc[2] = { MKF(30, 256, 128, true), MKF(52, 256, 128, true) };
    const float* attnW[2] = { P(40), P(62) };
    const float* attnB[2] = { P(41), P(63) };
    FwdW fDec = MKF(64, 128, 3, false);

    dim3 blk(64);
    int ntN = (N + 15) / 16, ntE = (E + 15) / 16;
    dim3 gN((ntN + WAVES - 1) / WAVES), gE((ntE + WAVES - 1) / WAVES);

    // encoders
    k_encode<13><<<gN, blk, 0, stream>>>((const float*)d_in[72], nodeLat, fNodeEnc, N);
    k_encode<6> <<<gE, blk, 0, stream>>>((const float*)d_in[73], edgeLat, fEdgeEnc, E);

    size_t zeroCount = (size_t)(((char*)sbuf + (size_t)N * 4) - (char*)agg) / 4;
    dim3 gLane((unsigned)(((size_t)E * 32 + 255) / 256));
    dim3 gEdge((E + 255) / 256);

    for (int p = 0; p < 2; ++p) {
        // alpha scores from OLD edge latents (before in-place edge_proc update)
        k_attn<<<gLane, dim3(256), 0, stream>>>(edgeLat, attnW[p], attnB[p], score, E);
        // edge update: edgeLat <- edgeLat + fwdnet([x_i, x_j, edgeLat])
        k_edge_proc<<<gE, blk, 0, stream>>>(srcIdx, dstIdx, nodeLat, edgeLat,
                                            fEdgeProc[p], E);
        // segment softmax over dst + weighted scatter of NEW edge latents
        k_zero<<<dim3(2048), dim3(256), 0, stream>>>(agg, zeroCount);
        k_smax<<<gEdge, dim3(256), 0, stream>>>(dstIdx, score, mbuf, E);
        k_expsum<<<gEdge, dim3(256), 0, stream>>>(dstIdx, score, mbuf, ex, sbuf, E);
        k_agg<<<gLane, dim3(256), 0, stream>>>(dstIdx, ex, sbuf, edgeLat, agg, E);
        // node update: nodeLat <- nodeLat + fwdnet([nodeLat, agg])
        k_node_proc<<<gN, blk, 0, stream>>>(nodeLat, agg, fNodeProc[p], N);
    }
    k_decoder<<<gN, blk, 0, stream>>>(nodeLat, (float*)d_out, fDec, N);
}
